// IPA_51591147160222
// MI455X (gfx1250) — matrix operand order repaired and checked by reading, not compiled, not run on hardware
//
#include <hip/hip_runtime.h>
#include <hip/hip_bf16.h>

typedef __bf16 bf16_t;
typedef __attribute__((ext_vector_type(16))) __bf16 v16bf;
typedef __attribute__((ext_vector_type(8)))  __bf16 v8bf;
typedef __attribute__((ext_vector_type(4)))  __bf16 v4bf;
typedef __attribute__((ext_vector_type(8)))  float  v8f;

// ---------------- problem dims ----------------
constexpr int  N    = 512;
constexpr int  CS   = 384;
constexpr int  CZ   = 128;
constexpr int  H    = 12;
constexpr int  C    = 16;
constexpr int  PV   = 8;
constexpr int  CAT  = 2017;
constexpr int  CATP = 2048;   // padded K for final GEMM
constexpr int  PROJ = 1152;   // 192*3 + 144*2 + 288
constexpr long NN   = (long)N * N;

// ---------------- workspace layout (bytes) ----------------
constexpr size_t OFF_ST   = 0;                      // bf16 s^T          512x384
constexpr size_t OFF_WCAT = OFF_ST   + 393216;      // bf16 W concat     1152x384
constexpr size_t OFF_WBB  = OFF_WCAT + 884736;      // bf16 Wb padded    16x128
constexpr size_t OFF_WSB  = OFF_WBB  + 4096;        // bf16 Ws padded    384x2048
constexpr size_t OFF_P    = OFF_WSB  + 1572864;     // f32  projections  1152x512
constexpr size_t OFF_QF   = OFF_P    + 2359296;     // bf16 q-feat       12x512x32 (f innermost)
constexpr size_t OFF_KF   = OFF_QF   + 393216;      // bf16 k-feat       12x512x32
constexpr size_t OFF_RN   = OFF_KF   + 393216;      // f32  -g|qg|^2     12x512
constexpr size_t OFF_CN   = OFF_RN   + 24576;       // f32  -g|kg|^2     12x512
constexpr size_t OFF_VV   = OFF_CN   + 24576;       // bf16 [v;vg] pad   12x48x512
constexpr size_t OFF_O1F  = OFF_VV   + 589824;      // f32  o1           1536x512
constexpr size_t OFF_O23  = OFF_O1F  + 3145728;     // f32  o2/o3g       12x48x512
constexpr size_t OFF_CATT = OFF_O23  + 1179648;     // bf16 cat^T        512x2048
constexpr size_t OFF_GH   = OFF_CATT + 2097152;     // f32  softplus(g)*wc/2, 16
constexpr size_t OFF_L    = OFF_GH   + 256;         // f32  b_bias       12x512x512
constexpr size_t OFF_AB   = OFF_L    + 12582912;    // bf16 attn         12x512x512
constexpr size_t OFF_ZB   = OFF_AB   + 6291456;     // bf16 z [c][i][j]  128x512x512
constexpr size_t OFF_ZT   = OFF_ZB   + 67108864;    // bf16 z [i][j][c]  512x512x128
// total ~158 MB

// ---------------- WMMA fragment helpers (wave32, 16x16x32 bf16, K contiguous) ----
// A (16x32, MxK row-major): lane<16 -> row=lane, elems K[0..15]
//                           lane>=16 -> row=lane-16, elems K[16..31]
__device__ __forceinline__ v16bf load_A_k1(const bf16_t* __restrict__ base, long ms,
                                           int lane) {
  int m  = lane & 15;
  int k0 = (lane >> 4) * 16;
  const bf16_t* p = base + (long)m * ms + k0;
  v8bf lo = *(const v8bf*)(p);        // 16B load
  v8bf hi = *(const v8bf*)(p + 8);    // 16B load
  return __builtin_shufflevector(lo, hi, 0, 1, 2, 3, 4, 5, 6, 7, 8, 9, 10, 11, 12, 13,
                                 14, 15);
}
// B (32x16, KxN), stored as N x K row-major (K contiguous per column)
// lane<16 -> col=lane, elems K[0..15]; lane>=16 -> col=lane-16, K[16..31]
__device__ __forceinline__ v16bf load_B_k1(const bf16_t* __restrict__ base, long ns,
                                           int lane) {
  int n  = lane & 15;
  int k0 = (lane >> 4) * 16;
  return *(const v16bf*)(base + (long)n * ns + k0);  // 32B -> 2x b128
}
// C/D (16x16 f32): lane<16 -> col=lane, M=vgpr; lane>=16 -> col=lane-16, M=vgpr+8
__device__ __forceinline__ void store_D(float* base, long ms, long ns, int lane,
                                        const v8f& d, int mmax) {
  int n    = lane & 15;
  int moff = (lane >> 4) * 8;
#pragma unroll
  for (int r = 0; r < 8; ++r) {
    int m = moff + r;
    if (m < mmax) base[(long)m * ms + (long)n * ns] = d[r];
  }
}
__device__ __forceinline__ v8f wmma_bf16(v16bf a, v16bf b, v8f c) {
  return __builtin_amdgcn_wmma_f32_16x16x32_bf16(false, a, false, b, (short)0, c,
                                                 false, false);
}

// ---------------- K0a: z -> bf16, both [c][i][j] and [i][j][c] layouts ----------------
__global__ void k_conv_z(const float* __restrict__ z, char* ws) {
  bf16_t* zB = (bf16_t*)(ws + OFF_ZB);
  bf16_t* zT = (bf16_t*)(ws + OFF_ZT);
  __shared__ bf16_t tile[CZ][68];  // 128 x 64 tile, padded vs bank conflicts
  int i  = blockIdx.x >> 3;        // 512 i
  int jb = (blockIdx.x & 7) * 64;  // 8 j-blocks
  int t = threadIdx.x;
  // phase 1: coalesced float4 reads of z[c][i][jb..jb+63]; write zB; stage LDS
#pragma unroll
  for (int q = 0; q < 8; ++q) {
    int idx = t + 256 * q;  // 0..2047 float4s
    int c   = idx >> 4;
    int jj  = (idx & 15) * 4;
    float4 v = *(const float4*)(z + (long)c * NN + (long)i * N + jb + jj);
    v4bf b;
    b[0] = (bf16_t)v.x; b[1] = (bf16_t)v.y; b[2] = (bf16_t)v.z; b[3] = (bf16_t)v.w;
    *(v4bf*)(zB + (long)c * NN + (long)i * N + jb + jj) = b;
    tile[c][jj + 0] = b[0]; tile[c][jj + 1] = b[1];
    tile[c][jj + 2] = b[2]; tile[c][jj + 3] = b[3];
  }
  __syncthreads();
  // phase 2: transposed, coalesced v4bf writes of zT[i][j][c]
#pragma unroll
  for (int q = 0; q < 8; ++q) {
    int lin = t + 256 * q;  // 0..2047
    int j   = lin >> 5;
    int c4  = (lin & 31) * 4;
    v4bf w;
    w[0] = tile[c4 + 0][j]; w[1] = tile[c4 + 1][j];
    w[2] = tile[c4 + 2][j]; w[3] = tile[c4 + 3][j];
    *(v4bf*)(zT + ((long)i * N + jb + j) * CZ + c4) = w;
  }
}

// ---------------- K0b: pack s^T, weights, gamma ----------------
__global__ void k_prep(const float* s, const float* Wq, const float* Wk, const float* Wv,
                       const float* Wqp, const float* Wkp, const float* Wvp,
                       const float* Wb, const float* Ws, const float* gamma, char* ws) {
  bf16_t* sT   = (bf16_t*)(ws + OFF_ST);
  bf16_t* Wcat = (bf16_t*)(ws + OFF_WCAT);
  bf16_t* WbB  = (bf16_t*)(ws + OFF_WBB);
  bf16_t* WsB  = (bf16_t*)(ws + OFF_WSB);
  float*  gh   = (float*)(ws + OFF_GH);
  const long T0 = (long)CS * N;
  const long T1 = T0 + (long)PROJ * CS;
  const long T2 = T1 + 16 * CZ;
  const long T3 = T2 + (long)CS * CATP;
  const long T4 = T3 + 16;
  long idx = (long)blockIdx.x * blockDim.x + threadIdx.x;
  long stride = (long)gridDim.x * blockDim.x;
  for (; idx < T4; idx += stride) {
    if (idx < T0) {
      int c = (int)(idx / N), n = (int)(idx % N);
      sT[(long)n * CS + c] = (bf16_t)s[idx];  // transpose: K (=c) innermost
    } else if (idx < T1) {
      long l = idx - T0; int r = (int)(l / CS), c = (int)(l % CS);
      float v;
      if      (r < 192) v = Wq [(long)r * CS + c];
      else if (r < 384) v = Wk [(long)(r - 192) * CS + c];
      else if (r < 576) v = Wv [(long)(r - 384) * CS + c];
      else if (r < 720) v = Wqp[(long)(r - 576) * CS + c];
      else if (r < 864) v = Wkp[(long)(r - 720) * CS + c];
      else              v = Wvp[(long)(r - 864) * CS + c];
      Wcat[l] = (bf16_t)v;
    } else if (idx < T2) {
      long l = idx - T1; int r = (int)(l / CZ), c = (int)(l % CZ);
      WbB[l] = (bf16_t)(r < H ? Wb[(long)r * CZ + c] : 0.0f);
    } else if (idx < T3) {
      long l = idx - T2; int r = (int)(l / CATP), c = (int)(l % CATP);
      WsB[l] = (bf16_t)(c < CAT ? Ws[(long)r * CAT + c] : 0.0f);
    } else {
      int h = (int)(idx - T3);
      // g_h = softplus(gamma_h) * w_c/2, w_c = sqrt(2/(9*P_Q))
      gh[h] = (h < H) ? log1pf(expf(gamma[h])) * 0.11785113f : 0.0f;
    }
  }
}

// ---------------- K1: P(1152x512) = Wcat(1152x384) x s(384x512), 4 m-tiles/wave ------
__global__ void k_proj(char* ws) {
  const bf16_t* Wcat = (const bf16_t*)(ws + OFF_WCAT);
  const bf16_t* sT   = (const bf16_t*)(ws + OFF_ST);
  float*        P    = (float*)(ws + OFF_P);
  int lane = threadIdx.x & 31, w = threadIdx.x >> 5;
  int tile = blockIdx.x * 8 + w;  // 18*32 wave-tiles (72 blocks)
  int mb = tile >> 5, nt = tile & 31;
  const bf16_t* Ab = Wcat + (long)mb * 64 * CS;
  const bf16_t* Bb = sT + (long)(nt * 16) * CS;
  v8f acc[4] = {};
  for (int kk = 0; kk < CS / 32; ++kk) {
    v16bf b = load_B_k1(Bb + kk * 32, CS, lane);
#pragma unroll
    for (int m = 0; m < 4; ++m) {
      v16bf a = load_A_k1(Ab + (long)m * 16 * CS + kk * 32, CS, lane);
      acc[m] = wmma_bf16(a, b, acc[m]);
    }
  }
#pragma unroll
  for (int m = 0; m < 4; ++m)
    store_D(P + ((long)mb * 64 + m * 16) * N + nt * 16, N, 1, lane, acc[m], 16);
}

// ---------------- K2: rigid transforms + feature packing (f innermost) ----------------
__global__ void k_xform(const float* __restrict__ t_r, const float* __restrict__ t_t,
                        char* ws) {
  const float* P  = (const float*)(ws + OFF_P);
  const float* gh = (const float*)(ws + OFF_GH);
  bf16_t* Qf = (bf16_t*)(ws + OFF_QF);  // [h][i][32]
  bf16_t* Kf = (bf16_t*)(ws + OFF_KF);  // [h][j][32]
  bf16_t* VV = (bf16_t*)(ws + OFF_VV);  // [h][48][512]
  float*  rn = (float*)(ws + OFF_RN);
  float*  cn = (float*)(ws + OFF_CN);
  int idx = blockIdx.x * blockDim.x + threadIdx.x;  // 12*512
  if (idx >= H * N) return;
  int h = idx / N, i = idx % N;
  const float* R = t_r + (long)i * 9;  // R[d*3+e] = t_r[i][d][e]
  const float* T = t_t + (long)i * 3;
  float g = gh[h];
  bf16_t* qrow = Qf + ((long)h * N + i) * 32;
  bf16_t* krow = Kf + ((long)h * N + i) * 32;
  for (int c = 0; c < C; ++c) {
    float qv = P[(long)(c * H + h) * N + i];
    float kv = P[(long)(192 + c * H + h) * N + i];
    float vv = P[(long)(384 + c * H + h) * N + i];
    qrow[c] = (bf16_t)(0.25f * qv);  // 1/sqrt(C)
    krow[c] = (bf16_t)kv;
    VV[((long)h * 48 + c) * N + i] = (bf16_t)vv;
  }
  float qn = 0.f, kn = 0.f;
  for (int p = 0; p < 4; ++p) {
    float ql[3], kl[3];
    for (int d = 0; d < 3; ++d) {
      ql[d] = P[(long)(576 + d * 48 + h * 4 + p) * N + i];
      kl[d] = P[(long)(720 + d * 48 + h * 4 + p) * N + i];
    }
    for (int d = 0; d < 3; ++d) {
      float qg = R[d * 3 + 0] * ql[0] + R[d * 3 + 1] * ql[1] + R[d * 3 + 2] * ql[2] + T[d];
      float kg = R[d * 3 + 0] * kl[0] + R[d * 3 + 1] * kl[1] + R[d * 3 + 2] * kl[2] + T[d];
      qn += qg * qg;
      kn += kg * kg;
      int f = 16 + d * 4 + p;
      qrow[f] = (bf16_t)(2.0f * g * qg);
      krow[f] = (bf16_t)kg;
    }
  }
  for (int f = 28; f < 32; ++f) { qrow[f] = (bf16_t)0.0f; krow[f] = (bf16_t)0.0f; }
  rn[h * N + i] = -g * qn;
  cn[h * N + i] = -g * kn;
  for (int p = 0; p < PV; ++p) {
    float vl[3];
    for (int d = 0; d < 3; ++d) vl[d] = P[(long)(864 + d * 96 + h * 8 + p) * N + i];
    for (int d = 0; d < 3; ++d) {
      float vg = R[d * 3 + 0] * vl[0] + R[d * 3 + 1] * vl[1] + R[d * 3 + 2] * vl[2] + T[d];
      VV[((long)h * 48 + 16 + d * 8 + p) * N + i] = (bf16_t)vg;
    }
  }
  for (int m = 40; m < 48; ++m) VV[((long)h * 48 + m) * N + i] = (bf16_t)0.0f;
}

// ---------------- K3: b_bias  L[h,i,j] = sum_c Wb[h,c] * z[c,i,j] ----------------
__global__ void k_bias(char* ws) {
  const bf16_t* WbB = (const bf16_t*)(ws + OFF_WBB);
  const bf16_t* zT  = (const bf16_t*)(ws + OFF_ZT);  // [i][j][c], c innermost
  float*        L   = (float*)(ws + OFF_L);
  int lane = threadIdx.x & 31, w = threadIdx.x >> 5;
  int tile = blockIdx.x * 8 + w;  // 512*32 tiles (2048 blocks); z read exactly once
  int i = tile >> 5, jt = tile & 31;
  const bf16_t* Bb = zT + ((long)i * N + jt * 16) * CZ;
  v8f acc = {};
#pragma unroll
  for (int kk = 0; kk < CZ / 32; ++kk) {
    v16bf a = load_A_k1(WbB + kk * 32, CZ, lane);
    v16bf b = load_B_k1(Bb + kk * 32, CZ, lane);
    acc = wmma_bf16(a, b, acc);
  }
  store_D(L + (long)i * N + jt * 16, NN, 1, lane, acc, H);  // m-stride = h stride
}

// ---------------- K4: logits (WMMA, K=32) + softmax over j ----------------
__global__ void k_attn(char* ws) {
  const bf16_t* Qf = (const bf16_t*)(ws + OFF_QF);
  const bf16_t* Kf = (const bf16_t*)(ws + OFF_KF);
  const float*  L  = (const float*)(ws + OFF_L);
  const float*  rn = (const float*)(ws + OFF_RN);
  const float*  cn = (const float*)(ws + OFF_CN);
  bf16_t*       aB = (bf16_t*)(ws + OFF_AB);
  __shared__ float Lt[16][512];
  __shared__ float tmp[256];
  __shared__ float rowred[16];
  const float w_l = 0.57735027f;  // sqrt(1/3)
  int h = blockIdx.x >> 5, it = blockIdx.x & 31, i0 = it * 16;
  int tid = threadIdx.x, lane = tid & 31, w = tid >> 5;
  // phase 1: each wave computes 4 logit tiles (uniform -> EXEC all ones for WMMA)
  v16bf a = load_A_k1(Qf + ((long)h * N + i0) * 32, 32, lane);
  for (int q = 0; q < 4; ++q) {
    int jt = w + q * 8;
    v16bf b = load_B_k1(Kf + ((long)h * N + jt * 16) * 32, 32, lane);
    v8f acc = {};
    acc = wmma_bf16(a, b, acc);
    int n = lane & 15, moff = (lane >> 4) * 8;
    int j = jt * 16 + n;
    float cnv = cn[h * N + j];
#pragma unroll
    for (int r = 0; r < 8; ++r) {
      int m = moff + r, i = i0 + m;
      Lt[m][j] = w_l * (acc[r] + L[(long)h * NN + (long)i * N + j] + rn[h * N + i] + cnv);
    }
  }
  __syncthreads();
  // phase 2: row-wise softmax (16 rows x 16 threads)
  int r = tid >> 4, c = tid & 15;
  float mx = -1e30f;
  for (int j = c; j < N; j += 16) mx = fmaxf(mx, Lt[r][j]);
  tmp[tid] = mx;
  __syncthreads();
  if (c == 0) {
    float m2 = tmp[r * 16];
    for (int t = 1; t < 16; ++t) m2 = fmaxf(m2, tmp[r * 16 + t]);
    rowred[r] = m2;
  }
  __syncthreads();
  float m2 = rowred[r];
  float sum = 0.f;
  for (int j = c; j < N; j += 16) {
    float e = __expf(Lt[r][j] - m2);
    Lt[r][j] = e;
    sum += e;
  }
  __syncthreads();
  tmp[tid] = sum;
  __syncthreads();
  if (c == 0) {
    float s2 = 0.f;
    for (int t = 0; t < 16; ++t) s2 += tmp[r * 16 + t];
    rowred[r] = s2;
  }
  __syncthreads();
  float inv = 1.0f / rowred[r];
  for (int j = c; j < N; j += 16)
    aB[(long)h * NN + (long)(i0 + r) * N + j] = (bf16_t)(Lt[r][j] * inv);
}

// ---------------- K5: O23[h] = [v;vg](48x512) x a[h]^T, 3 m-tiles/wave ----------------
__global__ void k_ov(char* ws) {
  const bf16_t* VV  = (const bf16_t*)(ws + OFF_VV);
  const bf16_t* aB  = (const bf16_t*)(ws + OFF_AB);
  float*        O23 = (float*)(ws + OFF_O23);
  int lane = threadIdx.x & 31, w = threadIdx.x >> 5;
  int tile = blockIdx.x * 8 + w;  // 12*32 wave-tiles (48 blocks)
  int h = tile >> 5, nt = tile & 31;
  const bf16_t* Ab = VV + (long)h * 48 * N;
  const bf16_t* Bb = aB + (long)h * NN + (long)(nt * 16) * N;  // B col n=i, K=j contiguous
  v8f acc[3] = {};
  for (int kk = 0; kk < N / 32; ++kk) {
    v16bf b = load_B_k1(Bb + kk * 32, N, lane);
#pragma unroll
    for (int m = 0; m < 3; ++m) {
      v16bf a = load_A_k1(Ab + (long)m * 16 * N + kk * 32, N, lane);
      acc[m] = wmma_bf16(a, b, acc[m]);
    }
  }
#pragma unroll
  for (int m = 0; m < 3; ++m)
    store_D(O23 + ((long)h * 48 + m * 16) * N + nt * 16, N, 1, lane, acc[m], 16);
}

// ---------------- K6: o1[c,h,i] = sum_j a[h,i,j] * z[c,i,j], 8 m-tiles/wave ----------
__global__ void k_o1(char* ws) {
  const bf16_t* zB  = (const bf16_t*)(ws + OFF_ZB);  // [c][i][j], j innermost
  const bf16_t* aB  = (const bf16_t*)(ws + OFF_AB);
  float*        o1f = (float*)(ws + OFF_O1F);
  int lane = threadIdx.x & 31, w = threadIdx.x >> 5;
  int i = blockIdx.x * 8 + w;  // one wave per i (64 blocks); full 128-row column block
  const bf16_t* Ab = zB + (long)i * N;  // A[m=c][k=j], m stride NN
  const bf16_t* Bb = aB + (long)i * N;  // B col n=h, K=j contiguous, n stride NN
  v8f acc[8] = {};
  for (int kk = 0; kk < N / 32; ++kk) {
    v16bf b = load_B_k1(Bb + kk * 32, NN, lane);  // a-fragment shared by all 8 m-tiles
#pragma unroll
    for (int m = 0; m < 8; ++m) {
      v16bf a = load_A_k1(Ab + (long)m * 16 * NN + kk * 32, NN, lane);
      acc[m] = wmma_bf16(a, b, acc[m]);
    }
  }
  int n = lane & 15, moff = (lane >> 4) * 8;
  if (n < H) {
#pragma unroll
    for (int m = 0; m < 8; ++m)
#pragma unroll
      for (int r = 0; r < 8; ++r) {
        int cc = m * 16 + moff + r;
        o1f[((long)cc * H + n) * N + i] = acc[m][r];  // row = c*H + h
      }
  }
}

// ---------------- K7: inverse transform + norm + assemble cat^T (bf16) --------------
__global__ void k_cat(const float* __restrict__ t_r, const float* __restrict__ t_t,
                      char* ws) {
  const float* o1f  = (const float*)(ws + OFF_O1F);
  const float* O23  = (const float*)(ws + OFF_O23);
  bf16_t*      catT = (bf16_t*)(ws + OFF_CATT);  // [i][r], r innermost
  __shared__ float red[256];
  int i = blockIdx.x, t = threadIdx.x;
  const float* R = t_r + (long)i * 9;  // R[e*3+d] = t_r[i][e][d]
  const float* T = t_t + (long)i * 3;
  bf16_t* crow = catT + (long)i * CATP;
  // o3 = R^T (o3g - t): rows 1728 + d*96 + h*8 + p ; accumulate squared norm
  float acc = 0.f;
  for (int r3 = t; r3 < 288; r3 += 256) {
    int d = r3 / 96, rem = r3 % 96, h = rem / 8, p = rem % 8;
    float x0 = O23[((long)h * 48 + 16 + 0 * 8 + p) * N + i] - T[0];
    float x1 = O23[((long)h * 48 + 16 + 1 * 8 + p) * N + i] - T[1];
    float x2 = O23[((long)h * 48 + 16 + 2 * 8 + p) * N + i] - T[2];
    float v = R[0 * 3 + d] * x0 + R[1 * 3 + d] * x1 + R[2 * 3 + d] * x2;
    crow[1728 + r3] = (bf16_t)v;
    acc += v * v;
  }
  red[t] = acc;
  __syncthreads();
  for (int s = 128; s > 0; s >>= 1) {
    if (t < s) red[t] += red[t + s];
    __syncthreads();
  }
  if (t == 0) crow[2016] = (bf16_t)sqrtf(red[0]);  // o3_norm row
  // o1 rows 0..1535, o2 rows 1536..1727, zero-pad rows 2017..2047
  for (int r = t; r < CATP; r += 256) {
    if (r < 1536) {
      crow[r] = (bf16_t)o1f[(long)r * N + i];
    } else if (r < 1728) {
      int r2 = r - 1536, cc = r2 / H, h = r2 % H;
      crow[r] = (bf16_t)O23[((long)h * 48 + cc) * N + i];
    } else if (r >= 2017) {
      crow[r] = (bf16_t)0.0f;
    }
  }
}

// ---------------- K8: out = Ws(384x2048) x cat(2048x512) + bs, 4 m-tiles/wave --------
__global__ void k_out(const float* __restrict__ bs, float* __restrict__ out, char* ws) {
  const bf16_t* WsB  = (const bf16_t*)(ws + OFF_WSB);
  const bf16_t* catT = (const bf16_t*)(ws + OFF_CATT);
  int lane = threadIdx.x & 31, w = threadIdx.x >> 5;
  int tile = blockIdx.x * 8 + w;  // 6*32 wave-tiles (24 blocks)
  int mb = tile >> 5, nt = tile & 31;
  const bf16_t* Ab = WsB + (long)mb * 64 * CATP;
  const bf16_t* Bb = catT + (long)(nt * 16) * CATP;
  v8f acc[4] = {};
  for (int kk = 0; kk < CATP / 32; ++kk) {
    v16bf b = load_B_k1(Bb + kk * 32, CATP, lane);
#pragma unroll
    for (int m = 0; m < 4; ++m) {
      v16bf a = load_A_k1(Ab + (long)m * 16 * CATP + kk * 32, CATP, lane);
      acc[m] = wmma_bf16(a, b, acc[m]);
    }
  }
  int n = lane & 15, moff = (lane >> 4) * 8;
#pragma unroll
  for (int m = 0; m < 4; ++m)
#pragma unroll
    for (int r = 0; r < 8; ++r) {
      int o = mb * 64 + m * 16 + moff + r;
      out[(long)o * N + nt * 16 + n] = acc[m][r] + bs[o];
    }
}

extern "C" void kernel_launch(void* const* d_in, const int* in_sizes, int n_in,
                              void* d_out, int out_size, void* d_ws, size_t ws_size,
                              hipStream_t stream) {
  (void)in_sizes; (void)n_in; (void)out_size; (void)ws_size;
  const float* s     = (const float*)d_in[0];
  const float* z     = (const float*)d_in[1];
  const float* t_r   = (const float*)d_in[2];
  const float* t_t   = (const float*)d_in[3];
  const float* Wq    = (const float*)d_in[4];
  const float* Wk    = (const float*)d_in[5];
  const float* Wv    = (const float*)d_in[6];
  const float* Wqp   = (const float*)d_in[7];
  const float* Wkp   = (const float*)d_in[8];
  const float* Wvp   = (const float*)d_in[9];
  const float* Wb    = (const float*)d_in[10];
  const float* gamma = (const float*)d_in[11];
  const float* Ws    = (const float*)d_in[12];
  const float* bs    = (const float*)d_in[13];
  char*  ws  = (char*)d_ws;
  float* out = (float*)d_out;

  k_conv_z<<<4096, 256, 0, stream>>>(z, ws);
  k_prep  <<<1024, 256, 0, stream>>>(s, Wq, Wk, Wv, Wqp, Wkp, Wvp, Wb, Ws, gamma, ws);
  k_proj  <<<72,   256, 0, stream>>>(ws);
  k_xform <<<24,   256, 0, stream>>>(t_r, t_t, ws);
  k_bias  <<<2048, 256, 0, stream>>>(ws);
  k_attn  <<<384,  256, 0, stream>>>(ws);
  k_ov    <<<48,   256, 0, stream>>>(ws);
  k_o1    <<<64,   256, 0, stream>>>(ws);
  k_cat   <<<512,  256, 0, stream>>>(t_r, t_t, ws);
  k_out   <<<24,   256, 0, stream>>>(bs, out, ws);
}
